// GIN_8546984919141
// MI455X (gfx1250) — matrix operand order repaired and checked by reading, not compiled, not run on hardware
//
#include <hip/hip_runtime.h>
#include <hip/hip_bf16.h>

typedef __attribute__((ext_vector_type(16))) _Float16 v16h;
typedef __attribute__((ext_vector_type(8)))  _Float16 v8h;
typedef __attribute__((ext_vector_type(8)))  float    v8f;
typedef __attribute__((ext_vector_type(4)))  float    v4f;

#define GIN_NODES   20000
#define GIN_EDGES   320000
#define GIN_GRAPHS  64
#define GIN_HID     256
#define GIN_IN      128
#define GIN_OUT     10

// ---------------------------------------------------------------------------
// Convert fp32 weight [K,N] (row-major, fin x fout) -> f16 transposed [N,K].
// This makes every WMMA B-fragment a single contiguous 32B v16h load per lane.
// ---------------------------------------------------------------------------
__global__ void gin_convert_wt(const float* __restrict__ W, _Float16* __restrict__ WT,
                               int K, int N) {
    int idx = blockIdx.x * blockDim.x + threadIdx.x;
    if (idx < K * N) {
        int k = idx / N, n = idx % N;
        WT[(size_t)n * K + k] = (_Float16)W[idx];
    }
}

// ---------------------------------------------------------------------------
// agg[dst[e]] += h[src[e]]   (one block per edge, one thread per feature)
// ---------------------------------------------------------------------------
__global__ void gin_scatter_add(const float* __restrict__ h, const int* __restrict__ src,
                                const int* __restrict__ dst, float* __restrict__ agg, int F) {
    int e = blockIdx.x;
    int f = threadIdx.x;
    int s = src[e], d = dst[e];
    atomicAdd(&agg[(size_t)d * F + f], h[(size_t)s * F + f]);
}

// ---------------------------------------------------------------------------
// WMMA GEMM: Out[M,256] = relu(A[M,K] @ W[K,256] + bias)
//   A_SUM_F32: A is (h + agg) fp32, converted to f16 in the A-fragment load
//   else     : A is f16
//   OUT_F16  : store f16 (feeds next GEMM) else fp32 (feeds next scatter)
// One 128-thread block (4 waves) per 16-row strip; each wave owns 16x64.
// ---------------------------------------------------------------------------
template <bool A_SUM_F32, bool OUT_F16>
__global__ __launch_bounds__(128)
void gin_gemm_wmma(const void* __restrict__ Aptr, const float* __restrict__ Agg,
                   const _Float16* __restrict__ WT, const float* __restrict__ bias,
                   void* __restrict__ Out, int K) {
    const int lane = threadIdx.x & 31;
    const int wave = threadIdx.x >> 5;
    const int rl   = lane & 15;      // A row / B,C column selector within tile
    const int hi   = lane >> 4;      // K-half selector (A,B), M-half selector (C)
    const int m0   = blockIdx.x * 16;
    const int n0   = wave * 64;
    const int m    = m0 + rl;

    v8f acc0 = {}, acc1 = {}, acc2 = {}, acc3 = {};

    for (int ko = 0; ko < K; ko += 32) {
        // ---- A fragment (contiguous k order k = 16*hi + elem, same as B) ----
        // lane 0-15 : elems 0..15 = K[ko..ko+15]
        // lane 16-31: elems 0..15 = K[ko+16..ko+31]
        v16h a;
        const int kA = ko + hi * 16;
        if constexpr (A_SUM_F32) {
            const float* hrow = (const float*)Aptr + (size_t)m * K + kA;
            const float* grow = Agg + (size_t)m * K + kA;
            v4f ha = *(const v4f*)(hrow);
            v4f hb = *(const v4f*)(hrow + 4);
            v4f hc = *(const v4f*)(hrow + 8);
            v4f hd = *(const v4f*)(hrow + 12);
            v4f ga = *(const v4f*)(grow);
            v4f gb = *(const v4f*)(grow + 4);
            v4f gc = *(const v4f*)(grow + 8);
            v4f gd = *(const v4f*)(grow + 12);
#pragma unroll
            for (int e = 0; e < 4; ++e) {
                a[e]      = (_Float16)(ha[e] + ga[e]);
                a[4 + e]  = (_Float16)(hb[e] + gb[e]);
                a[8 + e]  = (_Float16)(hc[e] + gc[e]);
                a[12 + e] = (_Float16)(hd[e] + gd[e]);
            }
        } else {
            const _Float16* arow = (const _Float16*)Aptr + (size_t)m * K + kA;
            v8h lo  = *(const v8h*)(arow);
            v8h hi8 = *(const v8h*)(arow + 8);
#pragma unroll
            for (int e = 0; e < 8; ++e) { a[e] = lo[e]; a[8 + e] = hi8[e]; }
        }

        // ---- B fragments: WT is [N,K] f16, so 16 consecutive K per lane ----
        // lanes 0-15: K[ko..ko+15] of column n ; lanes 16-31: K[ko+16..ko+31]
        const int kB = ko + hi * 16;
        const _Float16* wbase = WT + (size_t)(n0 + rl) * K + kB;
        v16h b0 = *(const v16h*)(wbase);
        v16h b1 = *(const v16h*)(wbase + 16 * (size_t)K);
        v16h b2 = *(const v16h*)(wbase + 32 * (size_t)K);
        v16h b3 = *(const v16h*)(wbase + 48 * (size_t)K);

        acc0 = __builtin_amdgcn_wmma_f32_16x16x32_f16(false, a, false, b0, (short)0, acc0, false, false);
        acc1 = __builtin_amdgcn_wmma_f32_16x16x32_f16(false, a, false, b1, (short)0, acc1, false, false);
        acc2 = __builtin_amdgcn_wmma_f32_16x16x32_f16(false, a, false, b2, (short)0, acc2, false, false);
        acc3 = __builtin_amdgcn_wmma_f32_16x16x32_f16(false, a, false, b3, (short)0, acc3, false, false);
    }

    // ---- epilogue: C layout lane0-15 N=lane / M=r, lane16-31 N=lane-16 / M=r+8
#pragma unroll
    for (int t = 0; t < 4; ++t) {
        v8f acc = (t == 0) ? acc0 : (t == 1) ? acc1 : (t == 2) ? acc2 : acc3;
        const int n  = n0 + t * 16 + rl;
        const float bv = bias[n];
#pragma unroll
        for (int r = 0; r < 8; ++r) {
            const int mm = m0 + hi * 8 + r;
            float v = acc[r] + bv;
            v = v > 0.0f ? v : 0.0f;
            if constexpr (OUT_F16)
                ((_Float16*)Out)[(size_t)mm * GIN_HID + n] = (_Float16)v;
            else
                ((float*)Out)[(size_t)mm * GIN_HID + n] = v;
        }
    }
}

// ---------------------------------------------------------------------------
// pooled[batch[node]] += h[node]
// ---------------------------------------------------------------------------
__global__ void gin_pool(const float* __restrict__ h, const int* __restrict__ batch,
                         float* __restrict__ pooled) {
    int node = blockIdx.x;
    int f = threadIdx.x;
    int g = batch[node];
    atomicAdd(&pooled[(size_t)g * GIN_HID + f], h[(size_t)node * GIN_HID + f]);
}

// ---------------------------------------------------------------------------
// logits = pooled @ wl + bl ; out = log_softmax(logits)  (one thread per graph)
// ---------------------------------------------------------------------------
__global__ void gin_classifier(const float* __restrict__ pooled, const float* __restrict__ wl,
                               const float* __restrict__ bl, float* __restrict__ out) {
    int g = threadIdx.x;
    if (g >= GIN_GRAPHS) return;
    float logits[GIN_OUT];
#pragma unroll
    for (int j = 0; j < GIN_OUT; ++j) logits[j] = bl[j];
    for (int k = 0; k < GIN_HID; ++k) {
        float p = pooled[g * GIN_HID + k];
#pragma unroll
        for (int j = 0; j < GIN_OUT; ++j) logits[j] += p * wl[k * GIN_OUT + j];
    }
    float mx = logits[0];
#pragma unroll
    for (int j = 1; j < GIN_OUT; ++j) mx = fmaxf(mx, logits[j]);
    float se = 0.0f;
#pragma unroll
    for (int j = 0; j < GIN_OUT; ++j) se += expf(logits[j] - mx);
    float lse = mx + logf(se);
#pragma unroll
    for (int j = 0; j < GIN_OUT; ++j) out[g * GIN_OUT + j] = logits[j] - lse;
}

extern "C" void kernel_launch(void* const* d_in, const int* in_sizes, int n_in,
                              void* d_out, int out_size, void* d_ws, size_t ws_size,
                              hipStream_t stream) {
    const float* x    = (const float*)d_in[0];
    const int*   eidx = (const int*)d_in[1];
    const int    E    = in_sizes[1] / 2;
    const int*   src  = eidx;
    const int*   dst  = eidx + E;
    const int*   batch = (const int*)d_in[2];
    const float* w1a = (const float*)d_in[3];  const float* b1a = (const float*)d_in[4];
    const float* w1b = (const float*)d_in[5];  const float* b1b = (const float*)d_in[6];
    const float* w2a = (const float*)d_in[7];  const float* b2a = (const float*)d_in[8];
    const float* w2b = (const float*)d_in[9];  const float* b2b = (const float*)d_in[10];
    const float* w3a = (const float*)d_in[11]; const float* b3a = (const float*)d_in[12];
    const float* w3b = (const float*)d_in[13]; const float* b3b = (const float*)d_in[14];
    const float* wl  = (const float*)d_in[15]; const float* bl  = (const float*)d_in[16];
    (void)n_in; (void)out_size;

    // -------- workspace layout (256B aligned chunks) --------
    char* ws = (char*)d_ws;
    size_t off = 0;
    auto alloc = [&](size_t bytes) -> void* {
        void* p = ws + off;
        off = (off + bytes + 255) & ~(size_t)255;
        return p;
    };
    float*    agg    = (float*)alloc((size_t)GIN_NODES * GIN_HID * 4);
    float*    hA     = (float*)alloc((size_t)GIN_NODES * GIN_HID * 4);
    float*    hB     = (float*)alloc((size_t)GIN_NODES * GIN_HID * 4);
    _Float16* t      = (_Float16*)alloc((size_t)GIN_NODES * GIN_HID * 2);
    _Float16* w1aT   = (_Float16*)alloc((size_t)GIN_IN  * GIN_HID * 2);
    _Float16* w1bT   = (_Float16*)alloc((size_t)GIN_HID * GIN_HID * 2);
    _Float16* w2aT   = (_Float16*)alloc((size_t)GIN_HID * GIN_HID * 2);
    _Float16* w2bT   = (_Float16*)alloc((size_t)GIN_HID * GIN_HID * 2);
    _Float16* w3aT   = (_Float16*)alloc((size_t)GIN_HID * GIN_HID * 2);
    _Float16* w3bT   = (_Float16*)alloc((size_t)GIN_HID * GIN_HID * 2);
    float*    pooled = (float*)alloc((size_t)GIN_GRAPHS * GIN_HID * 4);
    if (off > ws_size) return;  // workspace too small (deterministic no-op)

    const int mBlocks = GIN_NODES / 16;  // 1250, exact

    // -------- one-time weight conversion (fp32 [K,N] -> f16 [N,K]) --------
    gin_convert_wt<<<(GIN_IN * GIN_HID + 255) / 256, 256, 0, stream>>>(w1a, w1aT, GIN_IN,  GIN_HID);
    gin_convert_wt<<<(GIN_HID * GIN_HID + 255) / 256, 256, 0, stream>>>(w1b, w1bT, GIN_HID, GIN_HID);
    gin_convert_wt<<<(GIN_HID * GIN_HID + 255) / 256, 256, 0, stream>>>(w2a, w2aT, GIN_HID, GIN_HID);
    gin_convert_wt<<<(GIN_HID * GIN_HID + 255) / 256, 256, 0, stream>>>(w2b, w2bT, GIN_HID, GIN_HID);
    gin_convert_wt<<<(GIN_HID * GIN_HID + 255) / 256, 256, 0, stream>>>(w3a, w3aT, GIN_HID, GIN_HID);
    gin_convert_wt<<<(GIN_HID * GIN_HID + 255) / 256, 256, 0, stream>>>(w3b, w3bT, GIN_HID, GIN_HID);

    // -------- GIN layer 1 (input width 128) --------
    hipMemsetAsync(agg, 0, (size_t)GIN_NODES * GIN_IN * 4, stream);
    gin_scatter_add<<<E, GIN_IN, 0, stream>>>(x, src, dst, agg, GIN_IN);
    gin_gemm_wmma<true,  true ><<<mBlocks, 128, 0, stream>>>(x,  agg,     w1aT, b1a, t,  GIN_IN);
    gin_gemm_wmma<false, false><<<mBlocks, 128, 0, stream>>>(t,  nullptr, w1bT, b1b, hA, GIN_HID);

    // -------- GIN layer 2 --------
    hipMemsetAsync(agg, 0, (size_t)GIN_NODES * GIN_HID * 4, stream);
    gin_scatter_add<<<E, GIN_HID, 0, stream>>>(hA, src, dst, agg, GIN_HID);
    gin_gemm_wmma<true,  true ><<<mBlocks, 128, 0, stream>>>(hA, agg,     w2aT, b2a, t,  GIN_HID);
    gin_gemm_wmma<false, false><<<mBlocks, 128, 0, stream>>>(t,  nullptr, w2bT, b2b, hB, GIN_HID);

    // -------- GIN layer 3 --------
    hipMemsetAsync(agg, 0, (size_t)GIN_NODES * GIN_HID * 4, stream);
    gin_scatter_add<<<E, GIN_HID, 0, stream>>>(hB, src, dst, agg, GIN_HID);
    gin_gemm_wmma<true,  true ><<<mBlocks, 128, 0, stream>>>(hB, agg,     w3aT, b3a, t,  GIN_HID);
    gin_gemm_wmma<false, false><<<mBlocks, 128, 0, stream>>>(t,  nullptr, w3bT, b3b, hA, GIN_HID);

    // -------- pooling + classifier + log_softmax --------
    hipMemsetAsync(pooled, 0, (size_t)GIN_GRAPHS * GIN_HID * 4, stream);
    gin_pool<<<GIN_NODES, GIN_HID, 0, stream>>>(hA, batch, pooled);
    gin_classifier<<<1, 64, 0, stream>>>(pooled, wl, bl, (float*)d_out);
}
